// GraphEncoder_56573309223971
// MI455X (gfx1250) — compile-verified, matrix operand order checked by reading, not run on hardware
//
#include <hip/hip_runtime.h>

#define HIDC 128
#define LDS_STRIDE 136   // bf16 fragment rows: 272B -> 16B aligned, conflict-light
#define OUT_STRIDE 132   // f32 output tile rows: 528B -> 16B aligned, bank-spread
#define SCAN_CHUNK 1024

typedef __attribute__((ext_vector_type(16))) __bf16 v16bf;
typedef __attribute__((ext_vector_type(8)))  __bf16 v8bf;
typedef __attribute__((ext_vector_type(8)))  float  v8f;

// ---------------------------------------------------------------- utilities
__global__ void fill_i32(int* __restrict__ p, int v, int n) {
    int i = blockIdx.x * blockDim.x + threadIdx.x;
    int stride = gridDim.x * blockDim.x;
    for (; i < n; i += stride) p[i] = v;
}

__global__ void fill_f32(float* __restrict__ p, float v, int n) {
    int i = blockIdx.x * blockDim.x + threadIdx.x;
    int stride = gridDim.x * blockDim.x;
    for (; i < n; i += stride) p[i] = v;
}

__global__ void degree_kernel(const int* __restrict__ row, int* __restrict__ degI, int n_edges) {
    int i = blockIdx.x * blockDim.x + threadIdx.x;
    if (i < n_edges) atomicAdd(&degI[row[i]], 1);
}

__global__ void invsqrt_kernel(const int* __restrict__ degI, float* __restrict__ dis, int n) {
    int i = blockIdx.x * blockDim.x + threadIdx.x;
    if (i < n) {
        int d = degI[i];
        dis[i] = (d > 0) ? rsqrtf((float)d) : 0.0f;
    }
}

// ---------------------------------------------------------------- CSR build (scan)
// scan_partial: per-1024-chunk exclusive scan of degrees; chunk totals -> blockSums
__global__ __launch_bounds__(256)
void scan_partial(const int* __restrict__ degI, int* __restrict__ rowStart,
                  int* __restrict__ blockSums, int n)
{
    __shared__ int lds[256];
    const int t = threadIdx.x;
    const int base = blockIdx.x * SCAN_CHUNK;

    int v[4];
    int s = 0;
    #pragma unroll
    for (int i = 0; i < 4; ++i) {
        int idx = base + t * 4 + i;
        v[i] = s;                                   // thread-local exclusive prefix
        s += (idx < n) ? degI[idx] : 0;
    }
    lds[t] = s;
    __syncthreads();

    int x = s;                                      // Hillis-Steele inclusive scan
    for (int off = 1; off < 256; off <<= 1) {
        int y = (t >= off) ? lds[t - off] : 0;
        __syncthreads();
        x += y;
        lds[t] = x;
        __syncthreads();
    }
    const int excl = x - s;                         // exclusive prefix of thread sums

    #pragma unroll
    for (int i = 0; i < 4; ++i) {
        int idx = base + t * 4 + i;
        if (idx < n) rowStart[idx] = excl + v[i];
    }
    if (t == 255) blockSums[blockIdx.x] = x;        // chunk total
}

__global__ void scan_blocksums(int* __restrict__ blockSums, int nblocks) {
    if (threadIdx.x == 0 && blockIdx.x == 0) {
        int run = 0;
        for (int i = 0; i < nblocks; ++i) { int v = blockSums[i]; blockSums[i] = run; run += v; }
    }
}

__global__ void scan_addback(int* __restrict__ rowStart, const int* __restrict__ blockSums,
                             int* __restrict__ cursor, int n, int n_edges)
{
    int i = blockIdx.x * blockDim.x + threadIdx.x;
    if (i < n) {
        int v = rowStart[i] + blockSums[i / SCAN_CHUNK];
        rowStart[i] = v;
        cursor[i]   = v;
    }
    if (i == n) rowStart[n] = n_edges;
}

// bucket edges by destination row; precompute symmetric-norm weight per edge
__global__ void build_csr(const int* __restrict__ row, const int* __restrict__ col,
                          const float* __restrict__ dis, int* __restrict__ cursor,
                          int* __restrict__ edgeCol, float* __restrict__ edgeNorm, int n_edges)
{
    int e = blockIdx.x * blockDim.x + threadIdx.x;
    if (e < n_edges) {
        int r = row[e], c = col[e];
        int pos = atomicAdd(&cursor[r], 1);
        edgeCol[pos]  = c;
        edgeNorm[pos] = dis[r] * dis[c];
    }
}

// ---------------------------------------------------------------- WMMA GEMM
// Y[N,128] = (relu_in ? relu(X) : X) @ W[128,128] + bias[128]
// block = 256 threads (8 waves); block -> 16-row node tile; wave -> 16 cols.
__global__ __launch_bounds__(256)
void gemm_node(const float* __restrict__ X, const float* __restrict__ W,
               const float* __restrict__ bias, float* __restrict__ Y,
               int n_nodes, int relu_in)
{
    __shared__ __align__(16) __bf16 ldsA[16 * LDS_STRIDE];      // A tile [m][k]
    __shared__ __align__(16) __bf16 ldsBt[HIDC * LDS_STRIDE];   // W transposed [n][k]
    __shared__ __align__(16) float  ldsOut[16 * OUT_STRIDE];    // D tile for coalesced store

    const int tid  = threadIdx.x;
    const int wave = tid >> 5;
    const int lane = tid & 31;
    const int rowTile = blockIdx.x * 16;

    // Stage W (bf16, transposed to [n][k]) — global reads coalesced, L2-resident.
    for (int idx = tid; idx < HIDC * HIDC; idx += 256) {
        int k = idx >> 7;
        int n = idx & 127;
        ldsBt[n * LDS_STRIDE + k] = (__bf16)W[idx];
    }
    // Stage A tile (bf16, fused ReLU of previous layer).
    for (int idx = tid; idx < 16 * HIDC; idx += 256) {
        int m = idx >> 7;
        int k = idx & 127;
        float v = 0.0f;
        if (rowTile + m < n_nodes) v = X[(size_t)(rowTile + m) * HIDC + k];
        if (relu_in) v = v > 0.0f ? v : 0.0f;
        ldsA[m * LDS_STRIDE + k] = (__bf16)v;
    }
    __syncthreads();

    const int half = lane >> 4;       // 0: K 0..7/16..23 ; 1: K 8..15/24..31
    const int mn   = lane & 15;       // A row (m) / B,C column (n) within tile
    const int ncol = wave * 16 + mn;  // global output column

    // Two accumulator chains so dependent-WMMA hazard slots are filled.
    v8f acc0, acc1;
    const float bv = bias[ncol];
    #pragma unroll
    for (int r = 0; r < 8; ++r) { acc0[r] = bv; acc1[r] = 0.0f; }

    #pragma unroll
    for (int ks = 0; ks < HIDC; ks += 32) {
        v8bf a_lo = *(const v8bf*)&ldsA[mn * LDS_STRIDE + ks + 8 * half];
        v8bf a_hi = *(const v8bf*)&ldsA[mn * LDS_STRIDE + ks + 16 + 8 * half];
        v16bf a = __builtin_shufflevector(a_lo, a_hi,
                    0,1,2,3,4,5,6,7,8,9,10,11,12,13,14,15);
        v8bf b_lo = *(const v8bf*)&ldsBt[ncol * LDS_STRIDE + ks + 8 * half];
        v8bf b_hi = *(const v8bf*)&ldsBt[ncol * LDS_STRIDE + ks + 16 + 8 * half];
        v16bf b = __builtin_shufflevector(b_lo, b_hi,
                    0,1,2,3,4,5,6,7,8,9,10,11,12,13,14,15);

        if ((ks & 32) == 0)
            acc0 = __builtin_amdgcn_wmma_f32_16x16x32_bf16(
                       false, a, false, b, (short)0, acc0, false, false);
        else
            acc1 = __builtin_amdgcn_wmma_f32_16x16x32_bf16(
                       false, a, false, b, (short)0, acc1, false, false);
    }

    __syncthreads();   // all waves done reading A/B before output tile reuse

    // D layout: element r -> (m = r + 8*half, n = ncol); stage in LDS
    #pragma unroll
    for (int r = 0; r < 8; ++r) {
        int m = r + 8 * half;
        ldsOut[m * OUT_STRIDE + ncol] = acc0[r] + acc1[r];
    }
    __syncthreads();

    // Coalesced writeout: 2048 floats, 8 per thread (two b128 stores).
    {
        int m = tid >> 4;
        int k = (tid & 15) * 8;
        if (rowTile + m < n_nodes) {
            float4 o0 = *(const float4*)&ldsOut[m * OUT_STRIDE + k];
            float4 o1 = *(const float4*)&ldsOut[m * OUT_STRIDE + k + 4];
            *(float4*)&Y[(size_t)(rowTile + m) * HIDC + k]     = o0;
            *(float4*)&Y[(size_t)(rowTile + m) * HIDC + k + 4] = o1;
        }
    }
}

// ---------------------------------------------------------------- CSR aggregate
// One wave per node: lane owns 4 channels; register accumulation, no atomics.
__global__ __launch_bounds__(256)
void aggregate_csr(const float* __restrict__ h, const int* __restrict__ rowStart,
                   const int* __restrict__ edgeCol, const float* __restrict__ edgeNorm,
                   float* __restrict__ out, int n_nodes)
{
    const int gwave = (blockIdx.x * 256 + threadIdx.x) >> 5;
    const int lane  = threadIdx.x & 31;
    if (gwave >= n_nodes) return;

    const int beg = rowStart[gwave];
    const int end = rowStart[gwave + 1];

    float ax = 0.0f, ay = 0.0f, az = 0.0f, aw = 0.0f;
    int c = (beg < end) ? edgeCol[beg] : 0;
    for (int j = beg; j < end; ++j) {
        int cnext = (j + 1 < end) ? edgeCol[j + 1] : c;
        __builtin_prefetch(&h[(size_t)cnext * HIDC + lane * 4], 0, 1);
        float w = edgeNorm[j];
        float4 hv = *(const float4*)&h[(size_t)c * HIDC + lane * 4];
        ax += hv.x * w; ay += hv.y * w; az += hv.z * w; aw += hv.w * w;
        c = cnext;
    }
    float4 o; o.x = ax; o.y = ay; o.z = az; o.w = aw;
    *(float4*)&out[(size_t)gwave * HIDC + lane * 4] = o;
}

// ---------------------------------------------------------------- mean pool
__global__ __launch_bounds__(128)
void colsum_relu(const float* __restrict__ h, float* __restrict__ g, int n_nodes)
{
    const int cch = threadIdx.x;
    float s = 0.0f;
    for (int i = blockIdx.x; i < n_nodes; i += gridDim.x) {
        float v = h[(size_t)i * HIDC + cch];
        s += v > 0.0f ? v : 0.0f;
    }
    atomicAdd(&g[cch], s);
}

// ---------------------------------------------------------------- heads
__global__ __launch_bounds__(128)
void head_kernel(const float* __restrict__ g,
                 const float* __restrict__ Wmu, const float* __restrict__ bmu,
                 const float* __restrict__ Wlv, const float* __restrict__ blv,
                 float* __restrict__ out, float invN)
{
    const int j = threadIdx.x;   // 0..127 : [0,64) mu, [64,128) logvar
    if (j < 64) {
        float s = bmu[j];
        #pragma unroll 4
        for (int k = 0; k < HIDC; ++k) s += (g[k] * invN) * Wmu[k * 64 + j];
        out[j] = s;
    } else {
        int jj = j - 64;
        float s = blv[jj];
        #pragma unroll 4
        for (int k = 0; k < HIDC; ++k) s += (g[k] * invN) * Wlv[k * 64 + jj];
        out[j] = s;
    }
}

// ---------------------------------------------------------------- launch
extern "C" void kernel_launch(void* const* d_in, const int* in_sizes, int n_in,
                              void* d_out, int out_size, void* d_ws, size_t ws_size,
                              hipStream_t stream)
{
    const float* x   = (const float*)d_in[0];
    const int*   ei  = (const int*)  d_in[1];
    const float* W1  = (const float*)d_in[2];
    const float* b1  = (const float*)d_in[3];
    const float* W2  = (const float*)d_in[4];
    const float* b2  = (const float*)d_in[5];
    const float* Wmu = (const float*)d_in[6];
    const float* bmu = (const float*)d_in[7];
    const float* Wlv = (const float*)d_in[8];
    const float* blv = (const float*)d_in[9];

    const int n_nodes = in_sizes[0] / HIDC;
    const int n_edges = in_sizes[1] / 2;
    const int* row = ei;
    const int* col = ei + n_edges;

    // workspace layout (all 4-byte types; float4 users are 16B aligned)
    const size_t nfeat = (size_t)n_nodes * HIDC;
    float* bufA     = (float*)d_ws;                  // [N,128] linear outputs
    float* bufB     = bufA + nfeat;                  // [N,128] aggregated
    float* dis      = bufB + nfeat;                  // [N]
    int*   degI     = (int*)(dis + n_nodes);         // [N]
    int*   rowStart = degI + n_nodes;                // [N+1]
    int*   cursor   = rowStart + (n_nodes + 1);      // [N]
    int*   blockSums= cursor + n_nodes;              // [256]
    int*   edgeCol  = blockSums + 256;               // [E]
    float* edgeNorm = (float*)(edgeCol + n_edges);   // [E]
    float* g        = edgeNorm + n_edges;            // [128]

    const int gemmGrid = (n_nodes + 15) / 16;
    const int aggGrid  = (n_nodes * 32 + 255) / 256;         // 1 wave per node
    const int scanBlocks = (n_nodes + SCAN_CHUNK - 1) / SCAN_CHUNK;

    // ---- degrees, normalization, CSR build (once; reused by both convs)
    fill_i32<<<256, 256, 0, stream>>>(degI, 0, n_nodes);
    degree_kernel<<<(n_edges + 255) / 256, 256, 0, stream>>>(row, degI, n_edges);
    invsqrt_kernel<<<(n_nodes + 255) / 256, 256, 0, stream>>>(degI, dis, n_nodes);

    scan_partial<<<scanBlocks, 256, 0, stream>>>(degI, rowStart, blockSums, n_nodes);
    scan_blocksums<<<1, 64, 0, stream>>>(blockSums, scanBlocks);
    scan_addback<<<(n_nodes + 1 + 255) / 256, 256, 0, stream>>>(rowStart, blockSums,
                                                                cursor, n_nodes, n_edges);
    build_csr<<<(n_edges + 255) / 256, 256, 0, stream>>>(row, col, dis, cursor,
                                                         edgeCol, edgeNorm, n_edges);

    // ---- conv1: h1 = agg(x@W1+b1)
    gemm_node<<<gemmGrid, 256, 0, stream>>>(x, W1, b1, bufA, n_nodes, 0);
    aggregate_csr<<<aggGrid, 256, 0, stream>>>(bufA, rowStart, edgeCol, edgeNorm,
                                               bufB, n_nodes);

    // ---- conv2: h2 = agg(relu(h1)@W2+b2)   (ReLU fused into A staging)
    gemm_node<<<gemmGrid, 256, 0, stream>>>(bufB, W2, b2, bufA, n_nodes, 1);
    aggregate_csr<<<aggGrid, 256, 0, stream>>>(bufA, rowStart, edgeCol, edgeNorm,
                                               bufB, n_nodes);

    // ---- mean pool (ReLU fused) + heads
    fill_f32<<<1, 128, 0, stream>>>(g, 0.0f, HIDC);
    colsum_relu<<<1024, 128, 0, stream>>>(bufB, g, n_nodes);
    head_kernel<<<1, 128, 0, stream>>>(g, Wmu, bmu, Wlv, blv, (float*)d_out,
                                       1.0f / (float)n_nodes);
}
